// GraphLSTM_MTL_70007966925239
// MI455X (gfx1250) — compile-verified, matrix operand order checked by reading, not run on hardware
//
#include <hip/hip_runtime.h>

#define N_NODES 20000
#define T_STEPS 60
#define D_IN    20
#define H_DIM   128
#define G_DIM   128
#define HEADS   4
#define C_CH    32
#define E_EDGES 320000
#define EN      (E_EDGES + N_NODES)
#define FOURH   512
#define TILES16 (N_NODES / 16)   // 1250 row tiles of 16 (gemm kernel)
#define TILES32 (N_NODES / 32)   // 625 row tiles of 32 (lstm kernel)

// padded LDS row strides (elements) -> conflict-free banking
#define WHH_STR 136   // 68 dwords: bank step 4 per lane, 16 lanes distinct
#define WIH_STR 40    // 20 dwords: gcd(20,64)=4, 16 lanes distinct
#define HB_STR  136
#define CB_STR  132   // dwords (f32 rows)

typedef __attribute__((ext_vector_type(16))) __bf16 v16bf;
typedef __attribute__((ext_vector_type(8)))  float  v8f;

union Frag { unsigned u[8]; v16bf v; };

__device__ __forceinline__ unsigned short f2bf(float f) {
  unsigned u = __float_as_uint(f);
  u += 0x7FFFu + ((u >> 16) & 1u);          // round-to-nearest-even
  return (unsigned short)(u >> 16);
}
__device__ __forceinline__ unsigned pack_bf2(float lo, float hi) {
  return (unsigned)f2bf(lo) | ((unsigned)f2bf(hi) << 16);
}
__device__ __forceinline__ float sigmoid_f(float x) { return __frcp_rn(1.0f + __expf(-x)); }
__device__ __forceinline__ float tanh_f(float x) {
  float e = __expf(-2.0f * x);              // tanh = (1-e)/(1+e)
  return (1.0f - e) * __frcp_rn(1.0f + e);
}

// K-offset of fragment dword v for half-wave group grp (ISA 7.12.2 16-bit A/B layout)
__device__ __forceinline__ int k0_of(int v, int grp) {
  return ((v < 4) ? (2 * v) : (16 + 2 * (v - 4))) + 8 * grp;
}

// ---------------------------------------------------------------------------
// Phase 1: per-node LSTM. 4 waves/block, each wave owns a 32-row tile (2 WMMA
// M-tiles sharing every B fragment -> half the LDS B traffic per MAC).
// c kept in f32 LDS (padded), h in bf16 LDS, weights bf16 in padded LDS.
// ---------------------------------------------------------------------------
__global__ __launch_bounds__(128)
void lstm_kernel(const float* __restrict__ x, const float* __restrict__ Wih,
                 const float* __restrict__ Whh, const float* __restrict__ bih,
                 const float* __restrict__ bhh, float* __restrict__ node)
{
  extern __shared__ char smem[];
  unsigned short* whh_s = (unsigned short*)smem;            // [512][WHH_STR]
  unsigned short* wih_s = whh_s + FOURH * WHH_STR;          // [512][WIH_STR]
  unsigned short* hb    = wih_s + FOURH * WIH_STR;          // 4 x [32][HB_STR]
  float* bias_s = (float*)(hb + 4 * 32 * HB_STR);           // [512]
  float* cb     = bias_s + FOURH;                           // 4 x [32][CB_STR]

  const int tid = threadIdx.x;
  for (int i = tid; i < FOURH * H_DIM; i += 128) {
    int r = i >> 7, k = i & 127;
    whh_s[r * WHH_STR + k] = f2bf(Whh[i]);
  }
  for (int i = tid; i < FOURH * 32; i += 128) {
    int r = i >> 5, k = i & 31;
    wih_s[r * WIH_STR + k] = (k < D_IN) ? f2bf(Wih[r * D_IN + k]) : (unsigned short)0;
  }
  for (int i = tid; i < FOURH; i += 128) bias_s[i] = bih[i] + bhh[i];
  for (int i = tid; i < 4 * 32 * HB_STR; i += 128) hb[i] = 0;
  for (int i = tid; i < 4 * 32 * CB_STR; i += 128) cb[i] = 0.f;
  __syncthreads();

  const int wave = tid >> 5;
  const int lane = tid & 31;
  const int grp  = lane >> 4;
  const int l15  = lane & 15;            // A row index == B column index
  const int tile = blockIdx.x * 4 + wave;
  const bool active = tile < TILES32;    // wave-uniform -> EXEC all-ones inside
  const int row_base = tile * 32;
  unsigned short* myh = hb + wave * (32 * HB_STR);
  float*          myc = cb + wave * (32 * CB_STR);

  for (int t = 0; t < T_STEPS; ++t) {
    if (active) {
      // A fragments for x_t for both 16-row sub-tiles (K padded 20 -> 32)
      Frag ax[2];
      #pragma unroll
      for (int sub = 0; sub < 2; ++sub) {
        const float* xp = x + ((size_t)(row_base + sub * 16 + l15) * T_STEPS + t) * D_IN;
        if (t + 1 < T_STEPS) __builtin_prefetch(xp + D_IN, 0, 1);  // next x_t row
        #pragma unroll
        for (int v = 0; v < 8; ++v) {
          int k0 = k0_of(v, grp);
          ax[sub].u[v] = (k0 + 1 < D_IN) ? pack_bf2(xp[k0], xp[k0 + 1]) : 0u;
        }
      }
      // A fragments for h_{t-1}: 2 sub-tiles x 4 K-slabs
      Frag ah[2][4];
      #pragma unroll
      for (int sub = 0; sub < 2; ++sub)
        #pragma unroll
        for (int kk = 0; kk < 4; ++kk)
          #pragma unroll
          for (int v = 0; v < 8; ++v) {
            int k0 = k0_of(v, grp);
            ah[sub][kk].u[v] =
              *(const unsigned*)(myh + (sub * 16 + l15) * HB_STR + kk * 32 + k0);
          }

      #pragma unroll 1
      for (int j = 0; j < 8; ++j) {       // 8 column tiles of H
        v8f acc[4][2];
        #pragma unroll
        for (int g = 0; g < 4; ++g) {     // gates i,f,g,o
          const int colbase = g * H_DIM + j * 16;
          float b = bias_s[colbase + l15];
          v8f binit = { b, b, b, b, b, b, b, b };
          // shared Wih B fragment -> 2 WMMAs
          Frag bw;
          #pragma unroll
          for (int v = 0; v < 8; ++v) {
            int k0 = k0_of(v, grp);
            bw.u[v] = *(const unsigned*)(wih_s + (colbase + l15) * WIH_STR + k0);
          }
          #pragma unroll
          for (int sub = 0; sub < 2; ++sub)
            acc[g][sub] = __builtin_amdgcn_wmma_f32_16x16x32_bf16(
                false, ax[sub].v, false, bw.v, (short)0, binit, false, false);
          // shared Whh B fragments -> 2 WMMAs each
          #pragma unroll
          for (int kk = 0; kk < 4; ++kk) {
            Frag bh_;
            #pragma unroll
            for (int v = 0; v < 8; ++v) {
              int k0 = k0_of(v, grp);
              bh_.u[v] =
                *(const unsigned*)(whh_s + (colbase + l15) * WHH_STR + kk * 32 + k0);
            }
            #pragma unroll
            for (int sub = 0; sub < 2; ++sub)
              acc[g][sub] = __builtin_amdgcn_wmma_f32_16x16x32_bf16(
                  false, ah[sub][kk].v, false, bh_.v, (short)0, acc[g][sub], false, false);
          }
        }
        // element-wise gate math; D layout: VGPR r -> M = r + 8*grp, N = l15
        #pragma unroll
        for (int sub = 0; sub < 2; ++sub)
          #pragma unroll
          for (int r = 0; r < 8; ++r) {
            int m = sub * 16 + r + 8 * grp;
            float ig = sigmoid_f(acc[0][sub][r]);
            float fg = sigmoid_f(acc[1][sub][r]);
            float gg = tanh_f(acc[2][sub][r]);
            float og = sigmoid_f(acc[3][sub][r]);
            float cprev = myc[m * CB_STR + j * 16 + l15];
            float c = fg * cprev + ig * gg;
            myc[m * CB_STR + j * 16 + l15] = c;
            float hv = og * tanh_f(c);
            myh[m * HB_STR + j * 16 + l15] = f2bf(hv);
            if (t == T_STEPS - 1)
              node[(size_t)(row_base + m) * H_DIM + j * 16 + l15] = hv;
          }
      }
    }
    __syncthreads();
  }
}

// ---------------------------------------------------------------------------
// Phase 2: Y[20000,128] = X[20000,128] @ W[128,128]  (WMMA bf16, padded W^T LDS)
// ---------------------------------------------------------------------------
__global__ __launch_bounds__(256)
void gemm128_kernel(const float* __restrict__ X, const float* __restrict__ W,
                    float* __restrict__ Y)
{
  extern __shared__ char smem[];
  unsigned short* wt = (unsigned short*)smem;   // [n][k] transposed, [128][WHH_STR]
  const int tid = threadIdx.x;
  for (int i = tid; i < H_DIM * H_DIM; i += 256) {
    int nn = i >> 7, k = i & 127;
    wt[nn * WHH_STR + k] = f2bf(W[k * G_DIM + nn]);
  }
  __syncthreads();

  const int wave = tid >> 5, lane = tid & 31, grp = lane >> 4, l15 = lane & 15;
  const int tile = blockIdx.x * 8 + wave;
  if (tile >= TILES16) return;                  // wave-uniform, after the barrier
  const int row_base = tile * 16;

  Frag ax[4];
  const float* xp = X + (size_t)(row_base + l15) * H_DIM;
  #pragma unroll
  for (int kk = 0; kk < 4; ++kk)
    #pragma unroll
    for (int v = 0; v < 8; ++v) {
      int k0 = kk * 32 + k0_of(v, grp);
      ax[kk].u[v] = pack_bf2(xp[k0], xp[k0 + 1]);
    }

  for (int j = 0; j < 8; ++j) {
    v8f acc = { 0, 0, 0, 0, 0, 0, 0, 0 };
    #pragma unroll
    for (int kk = 0; kk < 4; ++kk) {
      Frag bw;
      #pragma unroll
      for (int v = 0; v < 8; ++v) {
        int k0 = k0_of(v, grp);
        bw.u[v] = *(const unsigned*)(wt + (j * 16 + l15) * WHH_STR + kk * 32 + k0);
      }
      acc = __builtin_amdgcn_wmma_f32_16x16x32_bf16(false, ax[kk].v, false, bw.v,
                                                    (short)0, acc, false, false);
    }
    #pragma unroll
    for (int r = 0; r < 8; ++r) {
      int m = r + 8 * grp;
      Y[(size_t)(row_base + m) * G_DIM + j * 16 + l15] = acc[r];
    }
  }
}

// ---------------------------------------------------------------------------
// Phase 3: GAT attention (gather/scatter, L2-resident)
// ---------------------------------------------------------------------------
__device__ __forceinline__ unsigned enc_f(float f) {
  unsigned u = __float_as_uint(f);
  return (u & 0x80000000u) ? ~u : (u | 0x80000000u);
}
__device__ __forceinline__ float dec_f(unsigned e) {
  unsigned u = (e & 0x80000000u) ? (e & 0x7FFFFFFFu) : ~e;
  return __uint_as_float(u);
}
__device__ __forceinline__ void edge_sd(const int* ei, int e, int& s, int& d) {
  if (e < E_EDGES) { s = ei[e]; d = ei[E_EDGES + e]; }
  else             { s = e - E_EDGES; d = s; }      // self loops
}
__device__ __forceinline__ float lrelu(float x) { return x > 0.f ? x : 0.2f * x; }

__global__ void attn_prep_kernel(const float* __restrict__ xh,
                                 const float* __restrict__ att_src,
                                 const float* __restrict__ att_dst,
                                 float* __restrict__ asr, float* __restrict__ adt)
{
  int idx = blockIdx.x * blockDim.x + threadIdx.x;
  if (idx >= N_NODES * HEADS) return;
  int nn = idx >> 2, hd = idx & 3;
  const float* xp = xh + (size_t)nn * G_DIM + hd * C_CH;
  const float* s = att_src + hd * C_CH;
  const float* d = att_dst + hd * C_CH;
  float a = 0.f, b = 0.f;
  #pragma unroll
  for (int c = 0; c < C_CH; ++c) { a += xp[c] * s[c]; b += xp[c] * d[c]; }
  asr[idx] = a; adt[idx] = b;
}

__global__ void zero_kernel(float* p, int n) {
  int i = blockIdx.x * blockDim.x + threadIdx.x;
  if (i < n) p[i] = 0.f;
}

__global__ void edge_max_kernel(const int* __restrict__ ei, const float* __restrict__ asr,
                                const float* __restrict__ adt, unsigned* __restrict__ emax)
{
  int idx = blockIdx.x * blockDim.x + threadIdx.x;
  if (idx >= EN * HEADS) return;
  int e = idx >> 2, hd = idx & 3, s, d;
  edge_sd(ei, e, s, d);
  float sc = lrelu(asr[s * HEADS + hd] + adt[d * HEADS + hd]);
  atomicMax(&emax[d * HEADS + hd], enc_f(sc));
}

__global__ void edge_sum_kernel(const int* __restrict__ ei, const float* __restrict__ asr,
                                const float* __restrict__ adt, const unsigned* __restrict__ emax,
                                float* __restrict__ denom)
{
  int idx = blockIdx.x * blockDim.x + threadIdx.x;
  if (idx >= EN * HEADS) return;
  int e = idx >> 2, hd = idx & 3, s, d;
  edge_sd(ei, e, s, d);
  float sc = lrelu(asr[s * HEADS + hd] + adt[d * HEADS + hd]);
  float m = dec_f(emax[d * HEADS + hd]);
  atomicAdd(&denom[d * HEADS + hd], __expf(sc - m));
}

__global__ void edge_agg_kernel(const int* __restrict__ ei, const float* __restrict__ asr,
                                const float* __restrict__ adt, const unsigned* __restrict__ emax,
                                const float* __restrict__ denom, const float* __restrict__ xh,
                                float* __restrict__ agg)
{
  int idx = blockIdx.x * blockDim.x + threadIdx.x;
  if (idx >= EN * HEADS) return;
  int e = idx >> 2, hd = idx & 3, s, d;
  edge_sd(ei, e, s, d);
  float sc = lrelu(asr[s * HEADS + hd] + adt[d * HEADS + hd]);
  float m  = dec_f(emax[d * HEADS + hd]);
  float w  = __expf(sc - m) / (denom[d * HEADS + hd] + 1e-16f);
  const float* xs = xh + (size_t)s * G_DIM + hd * C_CH;
  float* ag = agg + (size_t)d * G_DIM + hd * C_CH;
  #pragma unroll
  for (int c = 0; c < C_CH; ++c) atomicAdd(&ag[c], w * xs[c]);
}

__global__ void bias_elu_kernel(float* __restrict__ h, const float* __restrict__ b) {
  int i = blockIdx.x * blockDim.x + threadIdx.x;
  if (i >= N_NODES * G_DIM) return;
  float v = h[i] + b[i & 127];
  h[i] = v > 0.f ? v : (__expf(v) - 1.f);
}

__global__ void head_kernel(const float* __restrict__ h, const float* __restrict__ Wh,
                            const float* __restrict__ bh, float* __restrict__ out)
{
  int nn = blockIdx.x * blockDim.x + threadIdx.x;
  if (nn >= N_NODES) return;
  const float* hp = h + (size_t)nn * G_DIM;
  float o0 = bh[0], o1 = bh[1];
  #pragma unroll
  for (int k = 0; k < G_DIM; ++k) { float v = hp[k]; o0 += v * Wh[2 * k]; o1 += v * Wh[2 * k + 1]; }
  out[nn * 2 + 0] = o0;
  out[nn * 2 + 1] = o1;
}

// ---------------------------------------------------------------------------
extern "C" void kernel_launch(void* const* d_in, const int* in_sizes, int n_in,
                              void* d_out, int out_size, void* d_ws, size_t ws_size,
                              hipStream_t stream)
{
  (void)in_sizes; (void)n_in; (void)out_size; (void)ws_size;
  const float* x_win = (const float*)d_in[0];
  const int*   ei    = (const int*)d_in[1];
  const float* Wih   = (const float*)d_in[2];
  const float* Whh   = (const float*)d_in[3];
  const float* bih   = (const float*)d_in[4];
  const float* bhh   = (const float*)d_in[5];
  const float* W1    = (const float*)d_in[6];
  const float* as1   = (const float*)d_in[7];
  const float* ad1   = (const float*)d_in[8];
  const float* b1    = (const float*)d_in[9];
  const float* W2    = (const float*)d_in[10];
  const float* as2   = (const float*)d_in[11];
  const float* ad2   = (const float*)d_in[12];
  const float* b2    = (const float*)d_in[13];
  const float* Wh    = (const float*)d_in[14];
  const float* bh    = (const float*)d_in[15];
  float* out = (float*)d_out;

  float* buf0  = (float*)d_ws;                         // node -> agg/h
  float* buf1  = buf0 + (size_t)N_NODES * G_DIM;       // xh
  float* asr   = buf1 + (size_t)N_NODES * G_DIM;
  float* adt   = asr + N_NODES * HEADS;
  float* denom = adt + N_NODES * HEADS;
  unsigned* emax = (unsigned*)(denom + N_NODES * HEADS);

  // LSTM: 4 waves/block (one 32-row tile each), padded LDS ~278 KB of 320 KB
  const size_t lstm_lds =
      (size_t)(FOURH * WHH_STR + FOURH * WIH_STR + 4 * 32 * HB_STR) * 2   // bf16
    + (size_t)(FOURH + 4 * 32 * CB_STR) * 4;                              // f32
  lstm_kernel<<<(TILES32 + 3) / 4, 128, lstm_lds, stream>>>(x_win, Wih, Whh, bih, bhh, buf0);

  const int nblk16 = (TILES16 + 7) / 8;
  for (int layer = 0; layer < 2; ++layer) {
    const float* W  = layer ? W2  : W1;
    const float* as = layer ? as2 : as1;
    const float* ad = layer ? ad2 : ad1;
    const float* bb = layer ? b2  : b1;

    gemm128_kernel<<<nblk16, 256, (size_t)H_DIM * WHH_STR * 2, stream>>>(buf0, W, buf1);
    attn_prep_kernel<<<(N_NODES * HEADS + 255) / 256, 256, 0, stream>>>(buf1, as, ad, asr, adt);
    zero_kernel<<<(N_NODES * G_DIM + 255) / 256, 256, 0, stream>>>(buf0, N_NODES * G_DIM);
    zero_kernel<<<(N_NODES * HEADS + 255) / 256, 256, 0, stream>>>((float*)emax, N_NODES * HEADS);
    zero_kernel<<<(N_NODES * HEADS + 255) / 256, 256, 0, stream>>>(denom, N_NODES * HEADS);
    edge_max_kernel<<<(EN * HEADS + 255) / 256, 256, 0, stream>>>(ei, asr, adt, emax);
    edge_sum_kernel<<<(EN * HEADS + 255) / 256, 256, 0, stream>>>(ei, asr, adt, emax, denom);
    edge_agg_kernel<<<(EN * HEADS + 255) / 256, 256, 0, stream>>>(ei, asr, adt, emax, denom, buf1, buf0);
    bias_elu_kernel<<<(N_NODES * G_DIM + 255) / 256, 256, 0, stream>>>(buf0, bb);
  }
  head_kernel<<<(N_NODES + 255) / 256, 256, 0, stream>>>(buf0, Wh, bh, out);
}
